// Glm4MoeLiteMLPStack_28252294873723
// MI455X (gfx1250) — matrix operand order repaired and checked by reading, not compiled, not run on hardware
//
#include <hip/hip_runtime.h>
#include <hip/hip_bf16.h>

// MoE SwiGLU MLP stack (grouped GEMMs, tokens pre-sorted by expert) for gfx1250.
// fp32 -> bf16 on the staging path, f32 accumulation via v_wmma_f32_16x16x32_bf16.
// Memory-bound: ~2.24 GB HBM traffic => ~96 us floor @ 23.3 TB/s, provided
// (a) n-tiles iterate fastest so A-tiles and per-expert weight slabs stay L2
//     resident (192 MB L2), and
// (b) global loads for tile k+1 are issued before the WMMA block for tile k
//     (register-staged double buffer, ping-pong LDS).

typedef __bf16 bf16_t;
typedef __bf16 v16bf __attribute__((ext_vector_type(16)));
typedef float  v8f   __attribute__((ext_vector_type(8)));
typedef unsigned int u32x4 __attribute__((ext_vector_type(4)));
typedef unsigned int u32x2 __attribute__((ext_vector_type(2)));

union FragCvt { u32x4 q[2]; v16bf v; };   // two 16B LDS loads -> one WMMA operand

constexpr int KT = 32;   // K per WMMA (bf16)
constexpr int RS = 40;   // LDS row stride in bf16 elements (80B, 16B aligned, padded)

__device__ __forceinline__ unsigned pack_bf16(float a, float b) {
    union { __bf16 h; unsigned short u; } x, y;
    x.h = (__bf16)a; y.h = (__bf16)b;
    return (unsigned)x.u | ((unsigned)y.u << 16);
}

// Store one k-pair (2 rows of W) for 4 consecutive n columns, packed 2x bf16.
__device__ __forceinline__ void store_bpair(bf16_t* base, int n, int kp,
                                            float4 a, float4 b) {
    *reinterpret_cast<unsigned*>(&base[(n + 0) * RS + 2 * kp]) = pack_bf16(a.x, b.x);
    *reinterpret_cast<unsigned*>(&base[(n + 1) * RS + 2 * kp]) = pack_bf16(a.y, b.y);
    *reinterpret_cast<unsigned*>(&base[(n + 2) * RS + 2 * kp]) = pack_bf16(a.z, b.z);
    *reinterpret_cast<unsigned*>(&base[(n + 3) * RS + 2 * kp]) = pack_bf16(a.w, b.w);
}

// Tokens are sorted by expert; find which expert owns row m0.
__device__ __forceinline__ int find_expert(const int* __restrict__ gsz, int E, int m0) {
    int e = 0, off = 0;
    while (e < E) {
        int g = gsz[e];
        if (m0 >= off + g) { off += g; ++e; } else break;
    }
    return (e < E) ? e : (E - 1);
}

// ---------------------------------------------------------------------------
// Kernel 1: inter[T,F] = bf16( silu(x @ gate_w[e]) * (x @ up_w[e]) )
// Tile: 128 M x 128 N; 8 waves in 4(M) x 2(N); 32x64 per wave.
// blockIdx.x = n-tile (fast) so the A tile is L2-shared across n-blocks.
// ---------------------------------------------------------------------------
__global__ __launch_bounds__(256)
void moe_gateup_swiglu(const float* __restrict__ hs,
                       const int*   __restrict__ gsz,
                       const float* __restrict__ gate_w,
                       const float* __restrict__ up_w,
                       bf16_t*      __restrict__ inter,
                       int T, int H, int F, int E)
{
    __shared__ bf16_t As[2][128 * RS];   // A tile, row-major   [m][k]
    __shared__ bf16_t Bg[2][128 * RS];   // gate W tile, transposed [n][k]
    __shared__ bf16_t Bu[2][128 * RS];   // up   W tile, transposed [n][k]

    const int n0 = blockIdx.x * 128;
    const int m0 = blockIdx.y * 128;
    const int e  = find_expert(gsz, E, m0);
    const float* gw = gate_w + (size_t)e * H * F;
    const float* uw = up_w   + (size_t)e * H * F;

    const int tid  = threadIdx.x;
    const int wave = tid >> 5;
    const int lane = tid & 31;
    const int wm   = wave >> 1;          // 0..3 : M sub-block
    const int wn   = wave & 1;           // 0..1 : N sub-block
    const int l16  = lane & 15;
    const int hi   = (lane >> 4) & 1;    // half-wave selector for fragment layout

    v8f accg[2][4] = {};
    v8f accu[2][4] = {};

    // Register staging for the next K-tile (issued before the WMMA block).
    float4 ra[4];                        // A: 4 x float4 per thread
    float4 rg[2][2], ru[2][2];           // B: 2 jobs x (k-pair) per matrix

    auto load_stage = [&](int k0) {
#pragma unroll
        for (int j = 0; j < 4; ++j) {    // A: 128x32 fp32 = 1024 float4 chunks
            const int i   = tid + 256 * j;
            const int row = i >> 3;
            const int c4  = (i & 7) << 2;
            ra[j] = *reinterpret_cast<const float4*>(
                &hs[(size_t)(m0 + row) * H + k0 + c4]);
        }
#pragma unroll
        for (int j = 0; j < 2; ++j) {    // B: 16 k-pairs x 32 float4 cols = 512 jobs
            const int job = tid + 256 * j;
            const int kp  = job >> 5;
            const int c4  = (job & 31) << 2;
            const size_t go = (size_t)(k0 + 2 * kp) * F + n0 + c4;
            rg[j][0] = *reinterpret_cast<const float4*>(&gw[go]);
            rg[j][1] = *reinterpret_cast<const float4*>(&gw[go + F]);
            ru[j][0] = *reinterpret_cast<const float4*>(&uw[go]);
            ru[j][1] = *reinterpret_cast<const float4*>(&uw[go + F]);
        }
    };
    auto store_stage = [&](int buf) {
#pragma unroll
        for (int j = 0; j < 4; ++j) {
            const int i   = tid + 256 * j;
            const int row = i >> 3;
            const int c4  = (i & 7) << 2;
            u32x2 p = { pack_bf16(ra[j].x, ra[j].y), pack_bf16(ra[j].z, ra[j].w) };
            *reinterpret_cast<u32x2*>(&As[buf][row * RS + c4]) = p;
        }
#pragma unroll
        for (int j = 0; j < 2; ++j) {
            const int job = tid + 256 * j;
            const int kp  = job >> 5;
            const int c4  = (job & 31) << 2;
            store_bpair(Bg[buf], c4, kp, rg[j][0], rg[j][1]);
            store_bpair(Bu[buf], c4, kp, ru[j][0], ru[j][1]);
        }
    };

    load_stage(0);
    store_stage(0);
    int cur = 0;

    for (int k0 = 0; k0 < H; k0 += KT) {
        __syncthreads();                       // staging(cur) visible; prev compute done
        const bool has_next = (k0 + KT) < H;
        if (has_next) load_stage(k0 + KT);     // issue global loads early

        const bf16_t* Asb = As[cur];
        const bf16_t* Bgb = Bg[cur];
        const bf16_t* Bub = Bu[cur];

        FragCvt a[2];
#pragma unroll
        for (int mt = 0; mt < 2; ++mt) {
            const bf16_t* pa = &Asb[(wm * 32 + mt * 16 + l16) * RS + hi * 16];
            a[mt].q[0] = *reinterpret_cast<const u32x4*>(pa);
            a[mt].q[1] = *reinterpret_cast<const u32x4*>(pa + 8);
        }
#pragma unroll
        for (int nt = 0; nt < 4; ++nt) {
            const int col = wn * 64 + nt * 16 + l16;
            FragCvt bg, bu;
            const bf16_t* pg = &Bgb[col * RS + hi * 16];
            bg.q[0] = *reinterpret_cast<const u32x4*>(pg);
            bg.q[1] = *reinterpret_cast<const u32x4*>(pg + 8);
            const bf16_t* pu = &Bub[col * RS + hi * 16];
            bu.q[0] = *reinterpret_cast<const u32x4*>(pu);
            bu.q[1] = *reinterpret_cast<const u32x4*>(pu + 8);
#pragma unroll
            for (int mt = 0; mt < 2; ++mt) {
                accg[mt][nt] = __builtin_amdgcn_wmma_f32_16x16x32_bf16(
                    false, a[mt].v, false, bg.v, (short)0, accg[mt][nt], false, false);
                accu[mt][nt] = __builtin_amdgcn_wmma_f32_16x16x32_bf16(
                    false, a[mt].v, false, bu.v, (short)0, accu[mt][nt], false, false);
            }
        }

        if (has_next) store_stage(cur ^ 1);    // convert + LDS store after WMMAs
        cur ^= 1;
    }

    // ---- epilogue: SwiGLU, store bf16 intermediate ---------------------
#pragma unroll
    for (int mt = 0; mt < 2; ++mt) {
#pragma unroll
        for (int nt = 0; nt < 4; ++nt) {
            const int colg = n0 + wn * 64 + nt * 16 + l16;
#pragma unroll
            for (int r = 0; r < 8; ++r) {
                const int rowg = m0 + wm * 32 + mt * 16 + hi * 8 + r;
                const float g = accg[mt][nt][r];
                const float u = accu[mt][nt][r];
                const float s = g / (1.0f + __expf(-g));   // silu
                inter[(size_t)rowg * F + colg] = (bf16_t)(s * u);
            }
        }
    }
}

// ---------------------------------------------------------------------------
// Kernel 2: out[T,H] = inter[T,F] @ down_w[e][F,H]   (fp32 out)
// ---------------------------------------------------------------------------
__global__ __launch_bounds__(256)
void moe_down(const bf16_t* __restrict__ inter,
              const int*    __restrict__ gsz,
              const float*  __restrict__ down_w,
              float*        __restrict__ out,
              int T, int H, int F, int E)
{
    __shared__ bf16_t As[2][128 * RS];
    __shared__ bf16_t Bs[2][128 * RS];

    const int n0 = blockIdx.x * 128;
    const int m0 = blockIdx.y * 128;
    const int e  = find_expert(gsz, E, m0);
    const float* dw = down_w + (size_t)e * F * H;

    const int tid  = threadIdx.x;
    const int wave = tid >> 5;
    const int lane = tid & 31;
    const int wm   = wave >> 1;
    const int wn   = wave & 1;
    const int l16  = lane & 15;
    const int hi   = (lane >> 4) & 1;

    v8f acc[2][4] = {};

    u32x4  rA[2];                        // A: already bf16, 16B chunks
    float4 rB[2][2];                     // B: 2 jobs x (k-pair)

    auto load_stage = [&](int k0) {
#pragma unroll
        for (int j = 0; j < 2; ++j) {    // 128*32*2B / 16B = 512 chunks
            const int i   = tid + 256 * j;
            const int row = i >> 2;
            const int c   = (i & 3) << 3;
            rA[j] = *reinterpret_cast<const u32x4*>(
                &inter[(size_t)(m0 + row) * F + k0 + c]);
        }
#pragma unroll
        for (int j = 0; j < 2; ++j) {
            const int job = tid + 256 * j;
            const int kp  = job >> 5;
            const int c4  = (job & 31) << 2;
            const size_t go = (size_t)(k0 + 2 * kp) * H + n0 + c4;
            rB[j][0] = *reinterpret_cast<const float4*>(&dw[go]);
            rB[j][1] = *reinterpret_cast<const float4*>(&dw[go + H]);
        }
    };
    auto store_stage = [&](int buf) {
#pragma unroll
        for (int j = 0; j < 2; ++j) {
            const int i   = tid + 256 * j;
            const int row = i >> 2;
            const int c   = (i & 3) << 3;
            *reinterpret_cast<u32x4*>(&As[buf][row * RS + c]) = rA[j];
        }
#pragma unroll
        for (int j = 0; j < 2; ++j) {
            const int job = tid + 256 * j;
            const int kp  = job >> 5;
            const int c4  = (job & 31) << 2;
            store_bpair(Bs[buf], c4, kp, rB[j][0], rB[j][1]);
        }
    };

    load_stage(0);
    store_stage(0);
    int cur = 0;

    for (int k0 = 0; k0 < F; k0 += KT) {
        __syncthreads();
        const bool has_next = (k0 + KT) < F;
        if (has_next) load_stage(k0 + KT);

        const bf16_t* Asb = As[cur];
        const bf16_t* Bsb = Bs[cur];

        FragCvt a[2];
#pragma unroll
        for (int mt = 0; mt < 2; ++mt) {
            const bf16_t* pa = &Asb[(wm * 32 + mt * 16 + l16) * RS + hi * 16];
            a[mt].q[0] = *reinterpret_cast<const u32x4*>(pa);
            a[mt].q[1] = *reinterpret_cast<const u32x4*>(pa + 8);
        }
#pragma unroll
        for (int nt = 0; nt < 4; ++nt) {
            const int col = wn * 64 + nt * 16 + l16;
            FragCvt b;
            const bf16_t* pb = &Bsb[col * RS + hi * 16];
            b.q[0] = *reinterpret_cast<const u32x4*>(pb);
            b.q[1] = *reinterpret_cast<const u32x4*>(pb + 8);
#pragma unroll
            for (int mt = 0; mt < 2; ++mt) {
                acc[mt][nt] = __builtin_amdgcn_wmma_f32_16x16x32_bf16(
                    false, a[mt].v, false, b.v, (short)0, acc[mt][nt], false, false);
            }
        }

        if (has_next) store_stage(cur ^ 1);
        cur ^= 1;
    }

#pragma unroll
    for (int mt = 0; mt < 2; ++mt) {
#pragma unroll
        for (int nt = 0; nt < 4; ++nt) {
            const int colg = n0 + wn * 64 + nt * 16 + l16;
#pragma unroll
            for (int r = 0; r < 8; ++r) {
                const int rowg = m0 + wm * 32 + mt * 16 + hi * 8 + r;
                out[(size_t)rowg * H + colg] = acc[mt][nt][r];
            }
        }
    }
}

extern "C" void kernel_launch(void* const* d_in, const int* in_sizes, int n_in,
                              void* d_out, int out_size, void* d_ws, size_t ws_size,
                              hipStream_t stream) {
    (void)n_in; (void)out_size; (void)ws_size;
    const float* hs  = (const float*)d_in[0];   // [T,H] fp32
    const int*   gsz = (const int*)d_in[1];     // [E]
    const float* gw  = (const float*)d_in[2];   // [E,H,F]
    const float* uw  = (const float*)d_in[3];   // [E,H,F]
    const float* dw  = (const float*)d_in[4];   // [E,F,H]
    float*       out = (float*)d_out;           // [T,H] fp32

    constexpr int H = 2048;
    const int T = in_sizes[0] / H;
    const int E = in_sizes[1];
    const int F = (int)((long long)in_sizes[2] / ((long long)E * H));

    bf16_t* inter = (bf16_t*)d_ws;              // [T,F] bf16 scratch

    dim3 blk(256, 1, 1);
    dim3 g1(F / 128, T / 128, 1);               // n-tile fastest => L2 reuse
    dim3 g2(H / 128, T / 128, 1);
    hipLaunchKernelGGL(moe_gateup_swiglu, g1, blk, 0, stream,
                       hs, gsz, gw, uw, inter, T, H, F, E);
    hipLaunchKernelGGL(moe_down, g2, blk, 0, stream,
                       inter, gsz, dw, out, T, H, F, E);
}
